// MessagePassingNeuralNetwork_44667659879039
// MI455X (gfx1250) — matrix operand order repaired and checked by reading, not compiled, not run on hardware
//
#include <hip/hip_runtime.h>
#include <hip/hip_bf16.h>
#include <math.h>

typedef __attribute__((ext_vector_type(16))) _Float16 v16h;
typedef __attribute__((ext_vector_type(8)))  _Float16 v8h;
typedef __attribute__((ext_vector_type(8)))  float    v8f;

#define N_NODES   16000
#define N_EDGES   50000
#define NG        512
#define HDIM      64
#define DE        16
#define NLAYER    3
#define NS2S      3
#define K_CHUNKS  130              // 128 chunks of 32 cover K=4096 (W_e2), +2 for b_e2 (K=64, unit A)
#define E_TILES   (N_EDGES / 16)   // 3125
#define HT_ROW    72               // relu_t row padded: [64] = 1.0 (bias chunks), 16B-aligned stride

// ---------------- small dense kernels ----------------

__global__ void k_node_linear(const float* __restrict__ nf, const float* __restrict__ W,
                              const float* __restrict__ b, float* __restrict__ hx,
                              _Float16* __restrict__ lin16) {
  int idx = blockIdx.x * blockDim.x + threadIdx.x;
  if (idx >= N_NODES * HDIM) return;
  int n = idx >> 6, i = idx & 63;
  const float* x = nf + (size_t)n * HDIM;
  const float* w = W + (size_t)i * HDIM;
  float acc = b[i];
#pragma unroll 8
  for (int j = 0; j < HDIM; ++j) acc = fmaf(x[j], w[j], acc);
  hx[idx] = acc;
  lin16[idx] = (_Float16)acc;
}

__global__ void k_edge_hidden(const float* __restrict__ ef, const float* __restrict__ W,
                              const float* __restrict__ b, _Float16* __restrict__ ht16) {
  int idx = blockIdx.x * blockDim.x + threadIdx.x;
  if (idx >= N_EDGES * HDIM) return;
  int e = idx >> 6, i = idx & 63;
  const float* x = ef + (size_t)e * DE;
  const float* w = W + (size_t)i * DE;
  float acc = b[i];
#pragma unroll
  for (int k = 0; k < DE; ++k) acc = fmaf(x[k], w[k], acc);
  ht16[idx] = (_Float16)fmaxf(acc, 0.0f);
}

// Pre-swizzle B = permuted W_e2 (+ b_e2 tail) into WMMA fragment order:
// Bsw[cc*2048 + nt*512 + lane*16 + e] ; logical K_local for B lane element:
// oB = (lane>>4)*8 + (e&7) + (e>>3)*16 (table order, matches A fragment).
__global__ void k_prep_b(const float* __restrict__ W_e2, const float* __restrict__ b_e2,
                         _Float16* __restrict__ Bsw) {
  int idx = blockIdx.x * blockDim.x + threadIdx.x;
  if (idx >= K_CHUNKS * 2048) return;
  int e    = idx & 15;
  int lane = (idx >> 4) & 31;
  int nt   = (idx >> 9) & 3;
  int cc   = idx >> 11;
  int nOut = nt * 16 + (lane & 15);
  int oB   = (lane >> 4) * 8 + (e & 7) + ((e >> 3) << 4);
  int j    = ((cc & 1) << 5) + oB;
  float v;
  if (cc < 128) {
    int k = cc >> 1;
    v = W_e2[((size_t)(nOut * 64 + j)) * 64 + k];
  } else {
    v = b_e2[nOut * 64 + j];
  }
  Bsw[idx] = (_Float16)v;
}

__global__ void k_bounds(const int* __restrict__ n2g, int* __restrict__ sb, int* __restrict__ eb) {
  int i = blockIdx.x * blockDim.x + threadIdx.x;
  if (i >= N_NODES) return;
  int g = n2g[i];
  if (i == 0 || n2g[i - 1] != g) sb[g] = i;
  if (i == N_NODES - 1 || n2g[i + 1] != g) eb[g] = i + 1;
}

// ---------------- WMMA edge-message kernel ----------------
// Per wave: 16-edge tile, D[16x64] = A(16x4160) x B(4160x64),
// A[e, c] = relu_t[e, c>>6] * lin_src[e, c&63]  (bias chunks use padded relu_t[.,64]==1).
// Segment-sum fused via global f32 atomics into upd[dst].
__global__ void __launch_bounds__(128)
k_edge_msg(const _Float16* __restrict__ lin16, const _Float16* __restrict__ ht16,
           const _Float16* __restrict__ Bsw, const int* __restrict__ edge_list,
           float* __restrict__ upd) {
  __shared__ __align__(16) _Float16 lin_s[4][16][HDIM];
  __shared__ __align__(16) _Float16 ht_s[4][16][HT_ROW];
  int w    = threadIdx.x >> 5;
  int lane = threadIdx.x & 31;
  int tile = blockIdx.x * 4 + w;
  bool valid = tile < E_TILES;
  if (!valid) tile = E_TILES - 1;   // redundant work; atomics gated below
  int e0 = tile * 16;

  for (int r = 0; r < 16; ++r) {
    int e = e0 + r;
    int s = edge_list[2 * e];       // src node
    ((unsigned*)&lin_s[w][r][0])[lane] = ((const unsigned*)(lin16 + (size_t)s * HDIM))[lane];
    ((unsigned*)&ht_s [w][r][0])[lane] = ((const unsigned*)(ht16 + (size_t)e * HDIM))[lane];
  }
  if (lane < 16) ht_s[w][lane][64] = (_Float16)1.0f;   // unit scale for bias chunks
  __syncthreads();

  int m = lane & 15, half = lane >> 4;
  v8f acc0 = {}, acc1 = {}, acc2 = {}, acc3 = {};

  // Hoisted A row pointers: jbase toggles between 0 and 32.
  const _Float16* row0 = &lin_s[w][m][half * 8];        // jbase = 0
  const _Float16* row1 = &lin_s[w][m][32 + half * 8];   // jbase = 32
  v8h lo0 = *(const v8h*)(row0);
  v8h hi0 = *(const v8h*)(row0 + 16);
  v8h lo1 = *(const v8h*)(row1);
  v8h hi1 = *(const v8h*)(row1 + 16);

  const _Float16* bbase = Bsw + (size_t)lane * 16;
  v16h b0 = *(const v16h*)(bbase);
  v16h b1 = *(const v16h*)(bbase + 512);
  v16h b2 = *(const v16h*)(bbase + 1024);
  v16h b3 = *(const v16h*)(bbase + 1536);

  for (int cc = 0; cc < K_CHUNKS; ++cc) {
    // branchless scalar: bias chunks (cc=128,129) hit padded index 64 == 1.0
    _Float16 ak = ht_s[w][m][cc >> 1];
    v8h lo = (cc & 1) ? lo1 : lo0;
    v8h hi = (cc & 1) ? hi1 : hi0;
    v16h a;
#pragma unroll
    for (int i = 0; i < 8; ++i) { a[i] = lo[i] * ak; a[i + 8] = hi[i] * ak; }

    // prefetch next chunk's B fragments (Bsw has one extra dummy chunk)
    const _Float16* np = bbase + (size_t)(cc + 1) * 2048;
    v16h n0 = *(const v16h*)(np);
    v16h n1 = *(const v16h*)(np + 512);
    v16h n2 = *(const v16h*)(np + 1024);
    v16h n3 = *(const v16h*)(np + 1536);

    acc0 = __builtin_amdgcn_wmma_f32_16x16x32_f16(false, a, false, b0, (short)0, acc0, false, false);
    acc1 = __builtin_amdgcn_wmma_f32_16x16x32_f16(false, a, false, b1, (short)0, acc1, false, false);
    acc2 = __builtin_amdgcn_wmma_f32_16x16x32_f16(false, a, false, b2, (short)0, acc2, false, false);
    acc3 = __builtin_amdgcn_wmma_f32_16x16x32_f16(false, a, false, b3, (short)0, acc3, false, false);

    b0 = n0; b1 = n1; b2 = n2; b3 = n3;
  }
  if (!valid) return;

  // D layout: N = lane&15 ; M = vgpr + 8*half
  int n0i = lane & 15;
#pragma unroll
  for (int r = 0; r < 8; ++r) {
    int er = e0 + half * 8 + r;
    int d  = edge_list[2 * er + 1];   // dst node
    float* up = upd + (size_t)d * HDIM;
    atomicAdd(up + ( 0 + n0i), acc0[r]);
    atomicAdd(up + (16 + n0i), acc1[r]);
    atomicAdd(up + (32 + n0i), acc2[r]);
    atomicAdd(up + (48 + n0i), acc3[r]);
  }
}

// ---------------- GRU ----------------
__global__ void __launch_bounds__(256)
k_gru(const float* __restrict__ upd, float* __restrict__ hx, _Float16* __restrict__ lin16,
      const float* __restrict__ W_ih, const float* __restrict__ W_hh,
      const float* __restrict__ b_ih, const float* __restrict__ b_hh) {
  __shared__ float m_s[4][HDIM];
  __shared__ float h_s[4][HDIM];
  int ln = threadIdx.x >> 6, i = threadIdx.x & 63;
  int node = blockIdx.x * 4 + ln;
  if (node >= N_NODES) return;
  m_s[ln][i] = fmaxf(upd[(size_t)node * HDIM + i], 0.0f);
  h_s[ln][i] = hx[(size_t)node * HDIM + i];
  __syncthreads();
  const float* wr = W_ih + (size_t)i * HDIM;
  const float* wz = W_ih + (size_t)(64 + i) * HDIM;
  const float* wn = W_ih + (size_t)(128 + i) * HDIM;
  const float* ur = W_hh + (size_t)i * HDIM;
  const float* uz = W_hh + (size_t)(64 + i) * HDIM;
  const float* un = W_hh + (size_t)(128 + i) * HDIM;
  float xr = b_ih[i], xz = b_ih[64 + i], xn = b_ih[128 + i];
  float hr = b_hh[i], hz = b_hh[64 + i], hn = b_hh[128 + i];
  for (int j = 0; j < HDIM; ++j) {
    float mv = m_s[ln][j], hv = h_s[ln][j];
    xr = fmaf(mv, wr[j], xr); xz = fmaf(mv, wz[j], xz); xn = fmaf(mv, wn[j], xn);
    hr = fmaf(hv, ur[j], hr); hz = fmaf(hv, uz[j], hz); hn = fmaf(hv, un[j], hn);
  }
  float r = 1.0f / (1.0f + __expf(-(xr + hr)));
  float z = 1.0f / (1.0f + __expf(-(xz + hz)));
  float n = tanhf(xn + r * hn);
  float hnew = (1.0f - z) * n + z * h_s[ln][i];
  hx[(size_t)node * HDIM + i] = hnew;
  lin16[(size_t)node * HDIM + i] = (_Float16)hnew;
}

// ---------------- Set2Set ----------------
__global__ void __launch_bounds__(256)
k_lstm(const float* __restrict__ qstar, float* __restrict__ h, float* __restrict__ c,
       const float* __restrict__ W_lih, const float* __restrict__ W_lhh,
       const float* __restrict__ b_lih, const float* __restrict__ b_lhh) {
  __shared__ float q_s[2 * HDIM];
  __shared__ float h_s[HDIM];
  __shared__ float g_s[4 * HDIM];
  int g = blockIdx.x, t = threadIdx.x;
  if (t < 128) q_s[t] = qstar[(size_t)g * 128 + t];
  if (t < 64)  h_s[t] = h[(size_t)g * 64 + t];
  __syncthreads();
  const float* wi = W_lih + (size_t)t * 128;
  const float* wh = W_lhh + (size_t)t * 64;
  float acc = b_lih[t] + b_lhh[t];
  for (int j = 0; j < 128; ++j) acc = fmaf(q_s[j], wi[j], acc);
  for (int j = 0; j < 64;  ++j) acc = fmaf(h_s[j], wh[j], acc);
  g_s[t] = acc;
  __syncthreads();
  if (t < 64) {
    float ig = 1.0f / (1.0f + __expf(-g_s[t]));
    float fg = 1.0f / (1.0f + __expf(-g_s[64 + t]));
    float gg = tanhf(g_s[128 + t]);
    float og = 1.0f / (1.0f + __expf(-g_s[192 + t]));
    float cn = fg * c[(size_t)g * 64 + t] + ig * gg;
    float hn = og * tanhf(cn);
    c[(size_t)g * 64 + t] = cn;
    h[(size_t)g * 64 + t] = hn;
  }
}

__global__ void __launch_bounds__(256)
k_attn(const float* __restrict__ h, const float* __restrict__ nfeat,
       const int* __restrict__ sb, const int* __restrict__ eb,
       float* __restrict__ ebuf, float* __restrict__ qstar) {
  __shared__ float h_s[HDIM];
  __shared__ float red[256];
  __shared__ float outp[4][HDIM];
  int g = blockIdx.x, t = threadIdx.x;
  int s = sb[g], eN = eb[g];
  if (t < 64) h_s[t] = h[(size_t)g * 64 + t];
  __syncthreads();
  float lmax = -INFINITY;
  for (int i = s + t; i < eN; i += 256) {
    const float* x = nfeat + (size_t)i * HDIM;
    float p = 0.0f;
    for (int j = 0; j < HDIM; ++j) p = fmaf(h_s[j], x[j], p);
    ebuf[i] = p;
    lmax = fmaxf(lmax, p);
  }
  red[t] = lmax; __syncthreads();
  for (int o = 128; o > 0; o >>= 1) { if (t < o) red[t] = fmaxf(red[t], red[t + o]); __syncthreads(); }
  float mx = red[0];
  __syncthreads();
  float lsum = 0.0f;
  for (int i = s + t; i < eN; i += 256) {
    float ev = __expf(ebuf[i] - mx);
    ebuf[i] = ev;
    lsum += ev;
  }
  red[t] = lsum; __syncthreads();
  for (int o = 128; o > 0; o >>= 1) { if (t < o) red[t] += red[t + o]; __syncthreads(); }
  float ssum = red[0];
  float inv = (ssum > 0.0f) ? 1.0f / ssum : 0.0f;
  __syncthreads();
  int d = t & 63, part = t >> 6;
  float acc = 0.0f;
  for (int i = s + part; i < eN; i += 4) acc = fmaf(ebuf[i], nfeat[(size_t)i * HDIM + d], acc);
  outp[part][d] = acc;
  __syncthreads();
  if (t < 64) {
    float o = (outp[0][t] + outp[1][t] + outp[2][t] + outp[3][t]) * inv;
    qstar[(size_t)g * 128 + t] = h_s[t];
    qstar[(size_t)g * 128 + 64 + t] = o;
  }
}

__global__ void k_copy(const float* __restrict__ qstar, const float* __restrict__ hx,
                       float* __restrict__ out) {
  int i = blockIdx.x * blockDim.x + threadIdx.x;
  const int nq = NG * 128;
  const int total = nq + N_NODES * HDIM;
  if (i >= total) return;
  out[i] = (i < nq) ? qstar[i] : hx[i - nq];
}

// ---------------- launcher ----------------
extern "C" void kernel_launch(void* const* d_in, const int* in_sizes, int n_in,
                              void* d_out, int out_size, void* d_ws, size_t ws_size,
                              hipStream_t stream) {
  const float* node_feat = (const float*)d_in[0];
  const float* edge_feat = (const float*)d_in[1];
  const int*   edge_list = (const int*)d_in[2];
  const int*   node2graph= (const int*)d_in[3];
  const float* W_lin = (const float*)d_in[4];
  const float* b_lin = (const float*)d_in[5];
  const float* W_e1  = (const float*)d_in[6];
  const float* b_e1  = (const float*)d_in[7];
  const float* W_e2  = (const float*)d_in[8];
  const float* b_e2  = (const float*)d_in[9];
  const float* W_ih  = (const float*)d_in[10];
  const float* W_hh  = (const float*)d_in[11];
  const float* b_ih  = (const float*)d_in[12];
  const float* b_hh  = (const float*)d_in[13];
  const float* W_lih = (const float*)d_in[14];
  const float* W_lhh = (const float*)d_in[15];
  const float* b_lih = (const float*)d_in[16];
  const float* b_lhh = (const float*)d_in[17];

  char* p = (char*)d_ws;
  auto alloc = [&](size_t bytes) { char* r = p; p += (bytes + 255) & ~(size_t)255; return r; };
  float*    hx    = (float*)   alloc((size_t)N_NODES * HDIM * 4);
  float*    upd   = (float*)   alloc((size_t)N_NODES * HDIM * 4);
  _Float16* lin16 = (_Float16*)alloc((size_t)N_NODES * HDIM * 2);
  _Float16* ht16  = (_Float16*)alloc((size_t)N_EDGES * HDIM * 2);
  _Float16* Bsw   = (_Float16*)alloc((size_t)(K_CHUNKS + 1) * 2048 * 2); // +1 dummy chunk for prefetch
  float*    qstar = (float*)   alloc((size_t)NG * 128 * 4);
  float*    hbuf  = (float*)   alloc((size_t)NG * 64 * 4);
  float*    cbuf  = (float*)   alloc((size_t)NG * 64 * 4);
  int*      sbnd  = (int*)     alloc((size_t)NG * 4);
  int*      ebnd  = (int*)     alloc((size_t)NG * 4);
  float*    ebuf  = (float*)   alloc((size_t)N_NODES * 4);

  hipMemsetAsync(qstar, 0, (size_t)NG * 128 * 4, stream);
  hipMemsetAsync(hbuf,  0, (size_t)NG * 64 * 4, stream);
  hipMemsetAsync(cbuf,  0, (size_t)NG * 64 * 4, stream);
  hipMemsetAsync(sbnd,  0, (size_t)NG * 4, stream);
  hipMemsetAsync(ebnd,  0, (size_t)NG * 4, stream);

  k_node_linear<<<(N_NODES * HDIM + 255) / 256, 256, 0, stream>>>(node_feat, W_lin, b_lin, hx, lin16);
  k_edge_hidden<<<(N_EDGES * HDIM + 255) / 256, 256, 0, stream>>>(edge_feat, W_e1, b_e1, ht16);
  k_prep_b<<<(K_CHUNKS * 2048 + 255) / 256, 256, 0, stream>>>(W_e2, b_e2, Bsw);
  k_bounds<<<(N_NODES + 255) / 256, 256, 0, stream>>>(node2graph, sbnd, ebnd);

  for (int layer = 0; layer < NLAYER; ++layer) {
    hipMemsetAsync(upd, 0, (size_t)N_NODES * HDIM * 4, stream);
    k_edge_msg<<<(E_TILES + 3) / 4, 128, 0, stream>>>(lin16, ht16, Bsw, edge_list, upd);
    k_gru<<<N_NODES / 4, 256, 0, stream>>>(upd, hx, lin16, W_ih, W_hh, b_ih, b_hh);
  }

  for (int it = 0; it < NS2S; ++it) {
    k_lstm<<<NG, 256, 0, stream>>>(qstar, hbuf, cbuf, W_lih, W_lhh, b_lih, b_lhh);
    k_attn<<<NG, 256, 0, stream>>>(hbuf, hx, sbnd, ebnd, ebuf, qstar);
  }

  const int total = NG * 128 + N_NODES * HDIM;
  k_copy<<<(total + 255) / 256, 256, 0, stream>>>(qstar, hx, (float*)d_out);
}
